// DVGO_60413009986237
// MI455X (gfx1250) — matrix operand order repaired and checked by reading, not compiled, not run on hardware
//
#include <hip/hip_runtime.h>
#include <math.h>

// ---------------------------------------------------------------------------
// DVGO forward: trilinear grid sample + sin-encodings + 72->128->128->3 MLP.
// MLP runs on v_wmma_f32_16x16x32_f16 (f16 in, f32 accumulate).
// ---------------------------------------------------------------------------

typedef _Float16 half8  __attribute__((ext_vector_type(8)));
typedef _Float16 half16 __attribute__((ext_vector_type(16)));
typedef float    f32x8  __attribute__((ext_vector_type(8)));

#define SY    160
#define SX    25600      // 160*160
#define VOL   4096000    // 160^3
#define HSTR  136        // LDS feature-row stride in halfs (272B, bank-conflict pad)

// d_ws layout (bytes)
#define OFF_W1T 0        // 128 rows x 96  halfs (K padded 72->96)
#define OFF_W2T 24576    // 128 rows x 128 halfs
#define OFF_W3T 57344    // 16  rows x 128 halfs (N padded 3->16)
#define OFF_B1  61440    // 128 f32
#define OFF_B2  61952    // 128 f32
#define OFF_B3  62464    // 16  f32 (3 valid)

__device__ __forceinline__ half16 cat8(half8 lo, half8 hi) {
  return __builtin_shufflevector(lo, hi, 0,1,2,3,4,5,6,7,8,9,10,11,12,13,14,15);
}

// A fragment 16x32 f16 (ISA 7.12.2): lane = 16*hsel + m.
//   e in [0,8):  K = kc + 8*hsel + e
//   e in [8,16): K = kc + 16 + 8*hsel + (e-8)
__device__ __forceinline__ half16 loadA(const _Float16* rowp, int kc, int hsel) {
  half8 lo = *(const half8*)(rowp + kc + hsel * 8);
  half8 hi = *(const half8*)(rowp + kc + 16 + hsel * 8);
  return cat8(lo, hi);
}

// B fragment 32x16 f16: lane = 16*hsel + n ; same K order as A:
//   e in [0,8):  K = kc + 8*hsel + e
//   e in [8,16): K = kc + 16 + 8*hsel + (e-8)
__device__ __forceinline__ half16 loadB(const _Float16* wrow, int kc, int hsel) {
  half8 lo = *(const half8*)(wrow + kc + hsel * 8);
  half8 hi = *(const half8*)(wrow + kc + 16 + hsel * 8);
  return cat8(lo, hi);
}

__device__ __forceinline__ f32x8 wmma16(half16 a, half16 b, f32x8 c) {
  return __builtin_amdgcn_wmma_f32_16x16x32_f16(false, a, false, b, (short)0, c,
                                                false, false);
}

// ---------------------------------------------------------------------------
// Kernel 1: convert/transpose weights into d_ws (f16, K-contiguous rows).
// ---------------------------------------------------------------------------
__global__ void dvgo_prep(const float* __restrict__ W1, const float* __restrict__ b1,
                          const float* __restrict__ W2, const float* __restrict__ b2,
                          const float* __restrict__ W3, const float* __restrict__ b3,
                          _Float16* __restrict__ w1t, _Float16* __restrict__ w2t,
                          _Float16* __restrict__ w3t, float* __restrict__ b1f,
                          float* __restrict__ b2f, float* __restrict__ b3f) {
  int i      = blockIdx.x * blockDim.x + threadIdx.x;
  int stride = gridDim.x * blockDim.x;
  for (int idx = i; idx < 128 * 96; idx += stride) {   // W1: (72,128) -> [n][k<=96]
    int n = idx / 96, k = idx % 96;
    w1t[idx] = (k < 72) ? (_Float16)W1[k * 128 + n] : (_Float16)0.0f;
  }
  for (int idx = i; idx < 128 * 128; idx += stride) {  // W2: (128,128) -> [n][k]
    int n = idx / 128, k = idx % 128;
    w2t[idx] = (_Float16)W2[k * 128 + n];
  }
  for (int idx = i; idx < 16 * 128; idx += stride) {   // W3: (128,3) -> [n<16][k]
    int n = idx / 128, k = idx % 128;
    w3t[idx] = (n < 3) ? (_Float16)W3[k * 3 + n] : (_Float16)0.0f;
  }
  if (i < 128) { b1f[i] = b1[i]; b2f[i] = b2[i]; }
  if (i < 16)  { b3f[i] = (i < 3) ? b3[i] : 0.0f; }
}

// ---------------------------------------------------------------------------
// Kernel 2: main. 128 threads (4 waves) per block, 128 points per block.
// Phase 1: per-thread feature build + density. Phase 2: per-wave WMMA MLP.
// ---------------------------------------------------------------------------
__global__ __launch_bounds__(128)
void dvgo_main(const float* __restrict__ X,  const float* __restrict__ VD,
               const float* __restrict__ GD, const float* __restrict__ GK,
               const _Float16* __restrict__ W1T, const _Float16* __restrict__ W2T,
               const _Float16* __restrict__ W3T, const float* __restrict__ B1,
               const float* __restrict__ B2,    const float* __restrict__ B3,
               float* __restrict__ OUT, int n) {
  __shared__ _Float16 HS[128 * HSTR];   // per-point activation rows (f16)

  const int tid = threadIdx.x;
  const int p   = blockIdx.x * 128 + tid;
  _Float16* row = &HS[tid * HSTR];

  // ---------------- Phase 1: features + density ----------------
  if (p < n) {
    float px = X[3 * p], py = X[3 * p + 1], pz = X[3 * p + 2];
    float vx = VD[3 * p], vy = VD[3 * p + 1], vz = VD[3 * p + 2];

    float fx = px * 159.0f, fy = py * 159.0f, fz = pz * 159.0f;
    int ix = (int)floorf(fx); ix = ix < 0 ? 0 : (ix > 158 ? 158 : ix);
    int iy = (int)floorf(fy); iy = iy < 0 ? 0 : (iy > 158 ? 158 : iy);
    int iz = (int)floorf(fz); iz = iz < 0 ? 0 : (iz > 158 ? 158 : iz);
    float tx = fx - (float)ix, ty = fy - (float)iy, tz = fz - (float)iz;
    float ux = 1.0f - tx, uy = 1.0f - ty, uz = 1.0f - tz;

    const int o000 = ix * SX + iy * SY + iz;
    const int off[8] = {0, 1, SY, SY + 1, SX, SX + 1, SX + SY, SX + SY + 1};
    const float wgt[8] = {ux * uy * uz, ux * uy * tz, ux * ty * uz, ux * ty * tz,
                          tx * uy * uz, tx * uy * tz, tx * ty * uz, tx * ty * tz};

    // density (softplus, stable)
    float den = 0.0f;
    #pragma unroll
    for (int c = 0; c < 8; ++c) den += wgt[c] * GD[o000 + off[c]];
    float sp = (den > 0.0f) ? den + log1pf(expf(-den)) : log1pf(expf(den));
    OUT[3 * n + p] = sp;

    // k0 features (12 ch) -> cols 0..11
    #pragma unroll
    for (int ch = 0; ch < 12; ++ch) {
      const float* g = GK + ch * VOL + o000;
      float s = 0.0f;
      #pragma unroll
      for (int c = 0; c < 8; ++c) s += wgt[c] * g[off[c]];
      row[ch] = (_Float16)s;
    }
    // view encoding -> cols 12..38 : [v, sin(2^d v_i), cos(2^d v_i)]
    row[12] = (_Float16)vx; row[13] = (_Float16)vy; row[14] = (_Float16)vz;
    #pragma unroll
    for (int d = 0; d < 4; ++d) {
      float sc = (float)(1 << d);
      row[15 + d * 3 + 0] = (_Float16)sinf(vx * sc);
      row[15 + d * 3 + 1] = (_Float16)sinf(vy * sc);
      row[15 + d * 3 + 2] = (_Float16)sinf(vz * sc);
      row[27 + d * 3 + 0] = (_Float16)cosf(vx * sc);
      row[27 + d * 3 + 1] = (_Float16)cosf(vy * sc);
      row[27 + d * 3 + 2] = (_Float16)cosf(vz * sc);
    }
    // xyz encoding -> cols 39..71
    row[39] = (_Float16)px; row[40] = (_Float16)py; row[41] = (_Float16)pz;
    #pragma unroll
    for (int d = 0; d < 5; ++d) {
      float sc = (float)(1 << d);
      row[42 + d * 3 + 0] = (_Float16)sinf(px * sc);
      row[42 + d * 3 + 1] = (_Float16)sinf(py * sc);
      row[42 + d * 3 + 2] = (_Float16)sinf(pz * sc);
      row[57 + d * 3 + 0] = (_Float16)cosf(px * sc);
      row[57 + d * 3 + 1] = (_Float16)cosf(py * sc);
      row[57 + d * 3 + 2] = (_Float16)cosf(pz * sc);
    }
    #pragma unroll
    for (int k = 72; k < 96; ++k) row[k] = (_Float16)0.0f;  // K pad
  } else {
    for (int k = 0; k < 96; ++k) row[k] = (_Float16)0.0f;
  }
  __syncthreads();

  // ---------------- Phase 2: WMMA MLP ----------------
  const int lane  = tid & 31;
  const int wv    = tid >> 5;
  const int nlane = lane & 15;
  const int hsel  = lane >> 4;

  #pragma unroll
  for (int mt = 0; mt < 2; ++mt) {
    const int rowbase = wv * 32 + mt * 16;
    const _Float16* arow = &HS[(rowbase + nlane) * HSTR];

    // ---- Layer 1: 96 -> 128, ReLU ----
    half16 a0 = loadA(arow, 0, hsel);
    half16 a1 = loadA(arow, 32, hsel);
    half16 a2 = loadA(arow, 64, hsel);
    #pragma unroll
    for (int j = 0; j < 8; ++j) {
      float bv = B1[j * 16 + nlane];
      f32x8 c = {bv, bv, bv, bv, bv, bv, bv, bv};
      const _Float16* wrow = W1T + (j * 16 + nlane) * 96;
      c = wmma16(a0, loadB(wrow, 0, hsel), c);
      c = wmma16(a1, loadB(wrow, 32, hsel), c);
      c = wmma16(a2, loadB(wrow, 64, hsel), c);
      #pragma unroll
      for (int r = 0; r < 8; ++r)
        HS[(rowbase + r + 8 * hsel) * HSTR + j * 16 + nlane] =
            (_Float16)fmaxf(c[r], 0.0f);
    }

    // ---- Layer 2: 128 -> 128, ReLU ----
    half16 h0 = loadA(arow, 0, hsel);
    half16 h1 = loadA(arow, 32, hsel);
    half16 h2 = loadA(arow, 64, hsel);
    half16 h3 = loadA(arow, 96, hsel);
    #pragma unroll
    for (int j = 0; j < 8; ++j) {
      float bv = B2[j * 16 + nlane];
      f32x8 c = {bv, bv, bv, bv, bv, bv, bv, bv};
      const _Float16* wrow = W2T + (j * 16 + nlane) * 128;
      c = wmma16(h0, loadB(wrow, 0, hsel), c);
      c = wmma16(h1, loadB(wrow, 32, hsel), c);
      c = wmma16(h2, loadB(wrow, 64, hsel), c);
      c = wmma16(h3, loadB(wrow, 96, hsel), c);
      #pragma unroll
      for (int r = 0; r < 8; ++r)
        HS[(rowbase + r + 8 * hsel) * HSTR + j * 16 + nlane] =
            (_Float16)fmaxf(c[r], 0.0f);
    }

    // ---- Layer 3: 128 -> 3 (padded 16), sigmoid ----
    half16 g0 = loadA(arow, 0, hsel);
    half16 g1 = loadA(arow, 32, hsel);
    half16 g2 = loadA(arow, 64, hsel);
    half16 g3 = loadA(arow, 96, hsel);
    {
      float bv = B3[nlane];
      f32x8 c = {bv, bv, bv, bv, bv, bv, bv, bv};
      const _Float16* wrow = W3T + nlane * 128;
      c = wmma16(g0, loadB(wrow, 0, hsel), c);
      c = wmma16(g1, loadB(wrow, 32, hsel), c);
      c = wmma16(g2, loadB(wrow, 64, hsel), c);
      c = wmma16(g3, loadB(wrow, 96, hsel), c);
      #pragma unroll
      for (int r = 0; r < 8; ++r) {
        float sg = 1.0f / (1.0f + expf(-c[r]));
        int gp = blockIdx.x * 128 + rowbase + r + 8 * hsel;
        if (nlane < 3 && gp < n) OUT[gp * 3 + nlane] = sg;
      }
    }
  }
}

// ---------------------------------------------------------------------------
extern "C" void kernel_launch(void* const* d_in, const int* in_sizes, int n_in,
                              void* d_out, int out_size, void* d_ws, size_t ws_size,
                              hipStream_t stream) {
  const float* X  = (const float*)d_in[0];
  const float* VD = (const float*)d_in[1];
  const float* GD = (const float*)d_in[2];
  const float* GK = (const float*)d_in[3];
  const float* W1 = (const float*)d_in[4];
  const float* b1 = (const float*)d_in[5];
  const float* W2 = (const float*)d_in[6];
  const float* b2 = (const float*)d_in[7];
  const float* W3 = (const float*)d_in[8];
  const float* b3 = (const float*)d_in[9];
  float* out = (float*)d_out;

  char* ws = (char*)d_ws;
  _Float16* w1t = (_Float16*)(ws + OFF_W1T);
  _Float16* w2t = (_Float16*)(ws + OFF_W2T);
  _Float16* w3t = (_Float16*)(ws + OFF_W3T);
  float* b1f = (float*)(ws + OFF_B1);
  float* b2f = (float*)(ws + OFF_B2);
  float* b3f = (float*)(ws + OFF_B3);

  const int n = in_sizes[0] / 3;

  dvgo_prep<<<64, 256, 0, stream>>>(W1, b1, W2, b2, W3, b3,
                                    w1t, w2t, w3t, b1f, b2f, b3f);
  const int blocks = (n + 127) / 128;
  dvgo_main<<<blocks, 128, 0, stream>>>(X, VD, GD, GK, w1t, w2t, w3t,
                                        b1f, b2f, b3f, out, n);
}
